// SelfAttention_79877801771313
// MI455X (gfx1250) — compile-verified, matrix operand order checked by reading, not run on hardware
//
#include <hip/hip_runtime.h>
#include <stdint.h>

typedef __attribute__((ext_vector_type(16))) __bf16 v16bf;
typedef __attribute__((ext_vector_type(8)))  float  v8f;
typedef __attribute__((ext_vector_type(4)))  uint32_t v4u;
typedef __attribute__((ext_vector_type(8)))  int      v8i_;
typedef __attribute__((ext_vector_type(4)))  int      v4i_;
typedef uint16_t u16;
typedef uint32_t u32;

#define DEVI __device__ __forceinline__

constexpr int BATCH = 8;
constexpr int S     = 2048;
constexpr int DIN   = 1024;
constexpr int DE    = 1024;
constexpr int MTOT  = BATCH * S;

// ---------- helpers ----------
DEVI u16 f2bf(float f) {                       // round-to-nearest-even f32 -> bf16
    u32 u = __float_as_uint(f);
    u32 r = (u + 0x7FFFu + ((u >> 16) & 1u)) >> 16;
    return (u16)r;
}

struct U4x2 { uint4 a, b; };

DEVI v16bf ld_frag(const u16* p0, const u16* p1) {   // two 16B chunks -> 16 bf16 frag
    U4x2 u;
    u.a = *(const uint4*)p0;
    u.b = *(const uint4*)p1;
    return __builtin_bit_cast(v16bf, u);
}

DEVI v8f wmma_bf16(v16bf a, v16bf b, v8f c) {
    return __builtin_amdgcn_wmma_f32_16x16x32_bf16(
        /*neg_a=*/false, a, /*neg_b=*/false, b,
        /*c_mod=*/(short)0, c, /*reuse_a=*/false, /*reuse_b=*/false);
}

// TDM: load a 2D bf16 tile (rows x cols, row stride in elements) from global to
// LDS, inserting 16B of LDS padding after every 1024B (so an LDS "row" of
// 2048B data occupies 2080B -> bank-spread, 16B-aligned chunks).
DEVI void tdm_load_tile_2d(u32 lds_off, const u16* gptr,
                           int cols, int rows, int row_stride_elems) {
    uint64_t ga = (uint64_t)(uintptr_t)gptr;
    v4u g0 = { 1u,                                   // count = 1 descriptor
               lds_off,                              // lds_addr (bytes)
               (u32)ga,                              // global_addr[31:0]
               (u32)(ga >> 32) | (2u << 30) };       // global_addr[56:32] | type=2
    v8i_ g1;
    g1[0] = (int)((1u << 16)                          // data_size = 2 bytes
                | (1u << 20)                          // pad_enable
                | (7u << 22)                          // pad_interval = 256 dwords (1024B)
                | (3u << 25));                        // pad_amount  = 4 dwords (16B)
    g1[1] = (int)((u32)(cols & 0xFFFF) << 16);        // tensor_dim0 low16
    g1[2] = (int)(((u32)cols >> 16) | ((u32)(rows & 0xFFFF) << 16)); // dim0 hi | dim1 lo
    g1[3] = (int)(((u32)rows >> 16) | ((u32)cols << 16));            // dim1 hi | tile_dim0
    g1[4] = (int)(u32)rows;                           // tile_dim1 (tile_dim2 = 0)
    g1[5] = (int)(u32)row_stride_elems;               // tensor_dim0_stride low32
    g1[6] = 0;
    g1[7] = 0;
    v4i_ z4 = {};
#if __clang_major__ >= 23
    v8i_ z8 = {};
    __builtin_amdgcn_tensor_load_to_lds(g0, g1, z4, z4, z8, 0);
#else
    __builtin_amdgcn_tensor_load_to_lds(g0, g1, z4, z4, 0);
#endif
}

// ---------- f32 -> bf16 conversion (vectorized x4) ----------
__global__ void cvt_f32_bf16(const float* __restrict__ in, u16* __restrict__ out, int n4) {
    int i = blockIdx.x * blockDim.x + threadIdx.x;
    if (i < n4) {
        float4 f = ((const float4*)in)[i];
        ushort4 o;
        o.x = f2bf(f.x); o.y = f2bf(f.y); o.z = f2bf(f.z); o.w = f2bf(f.w);
        ((ushort4*)out)[i] = o;
    }
}

// ---------- f32 [K,N] -> bf16 transposed [N,K] (weights, done once) ----------
__global__ void cvt_f32_bf16_T(const float* __restrict__ in, u16* __restrict__ out,
                               int Kd, int N) {
    int idx = blockIdx.x * blockDim.x + threadIdx.x;
    if (idx < Kd * N) {
        int k = idx / N, n = idx - k * N;          // coalesced read along n
        out[(size_t)n * Kd + k] = f2bf(in[idx]);
    }
}

// ---------- tiled bf16 GEMM: C[M,N] = A[M,K] * Bt[N,K]^T + bias ----------
// A row-major [M,K]; Bt is the PRE-TRANSPOSED weight [N,K] (k contiguous).
// MODE 0: bf16 row-major out; MODE 1: bf16 out transposed per batch [b][n][s];
// MODE 2: f32 row-major out.
template<int MODE>
__global__ __launch_bounds__(256)
void gemm_bf16_k(const u16* __restrict__ A, const u16* __restrict__ Bt,
                 const float* __restrict__ bias, void* __restrict__ Out,
                 int M, int N, int Kd)
{
    constexpr int BM = 128, BN = 128, BK = 32;
    constexpr int LDA = 40;   // padded LDS row stride (shorts): 80B, 16B aligned
    constexpr int LDB = 40;
    __shared__ u16 As[BM * LDA];
    __shared__ u16 Bs[BN * LDB];   // [n][k]

    const int nb = N / BN;
    const int m0 = (blockIdx.x / nb) * BM;
    const int n0 = (blockIdx.x % nb) * BN;

    const int t    = threadIdx.x;
    const int lane = t & 31;
    const int wave = t >> 5;       // 0..7
    const int wm   = wave >> 2;    // 0..1 : 64-row slice
    const int wn   = wave & 3;     // 0..3 : 32-col slice
    const int half = lane >> 4;
    const int lr   = lane & 15;

    v8f acc[4][2] = {};

    for (int k0 = 0; k0 < Kd; k0 += BK) {
        // A tile: 128 rows x 32 cols = 512 x 16B chunks, 2 per thread
        #pragma unroll
        for (int i = 0; i < 2; i++) {
            int idx = t + i * 256;
            int row = idx >> 2, ch = idx & 3;
            uint4 d = *(const uint4*)(A + (size_t)(m0 + row) * Kd + k0 + ch * 8);
            *(uint4*)(&As[row * LDA + ch * 8]) = d;
        }
        // B tile from pre-transposed weights: plain copy, no scalar transpose
        #pragma unroll
        for (int i = 0; i < 2; i++) {
            int idx = t + i * 256;
            int row = idx >> 2, ch = idx & 3;
            uint4 d = *(const uint4*)(Bt + (size_t)(n0 + row) * Kd + k0 + ch * 8);
            *(uint4*)(&Bs[row * LDB + ch * 8]) = d;
        }
        // prefetch next k-tile into caches
        if (k0 + BK < Kd) {
            int row = t >> 2, ch = t & 3;
            __builtin_prefetch(A  + (size_t)(m0 + row) * Kd + k0 + BK + ch * 8, 0, 1);
            __builtin_prefetch(Bt + (size_t)(n0 + row) * Kd + k0 + BK + ch * 8, 0, 1);
        }
        __syncthreads();

        v16bf afr[4], bfr[2];
        #pragma unroll
        for (int i = 0; i < 4; i++) {
            const u16* rp = &As[(wm * 64 + i * 16 + lr) * LDA];
            afr[i] = ld_frag(rp + half * 8, rp + 16 + half * 8);
        }
        #pragma unroll
        for (int j = 0; j < 2; j++) {
            const u16* rp = &Bs[(wn * 32 + j * 16 + lr) * LDB];
            bfr[j] = ld_frag(rp + half * 8, rp + 16 + half * 8);
        }
        #pragma unroll
        for (int i = 0; i < 4; i++)
            #pragma unroll
            for (int j = 0; j < 2; j++)
                acc[i][j] = wmma_bf16(afr[i], bfr[j], acc[i][j]);
        __syncthreads();
    }

    // epilogue: lane lr = col, VGPR r -> row r + 8*half
    #pragma unroll
    for (int i = 0; i < 4; i++) {
        #pragma unroll
        for (int j = 0; j < 2; j++) {
            const int n = n0 + wn * 32 + j * 16 + lr;
            const float bv = bias[n];
            #pragma unroll
            for (int r = 0; r < 8; r++) {
                const int m = m0 + wm * 64 + i * 16 + r + half * 8;
                const float v = acc[i][j][r] + bv;
                if constexpr (MODE == 0) {
                    ((u16*)Out)[(size_t)m * N + n] = f2bf(v);
                } else if constexpr (MODE == 1) {
                    const int bb = m >> 11;          // m / S (S=2048)
                    const int ss = m & (S - 1);
                    ((u16*)Out)[((size_t)bb * N + n) * S + ss] = f2bf(v);
                } else {
                    ((float*)Out)[(size_t)m * N + n] = v;
                }
            }
        }
    }
}

// ---------- flash-style attention: ctx = softmax(Q K^T / sqrt(D)) V ----------
// One block handles 16 q-rows of one batch. 8 waves; each wave owns a 128-wide
// d-slice of the output accumulator. K is [b][s][d] bf16, Vt is [b][d][s] bf16.
// Q block is staged into LDS by the Tensor Data Mover (with 16B pad / 1024B).
__global__ __launch_bounds__(256)
void attn_k(const u16* __restrict__ Q, const u16* __restrict__ K,
            const u16* __restrict__ Vt, u16* __restrict__ ctx)
{
    constexpr int QB = 16;           // q rows per block
    constexpr int KC = 128;          // k rows per outer iteration (16 per wave)
    constexpr int LQ = DIN + 16;     // TDM-padded Qs row stride (shorts): 2080B
    constexpr int LP = KC + 8;       // padded Pbuf stride (shorts)

    __shared__ u16   Qs[QB * LQ];       // 33,280 B (TDM destination)
    __shared__ float Sbuf[QB][KC];      // 8 KB
    __shared__ u16   Pbuf[QB * LP];     // ~4.3 KB
    __shared__ float rowM[QB], rowL[QB], rowA[QB];

    const int qblk = blockIdx.x & (S / QB - 1);
    const int b    = blockIdx.x / (S / QB);
    const int q0   = qblk * QB;

    const int t    = threadIdx.x;
    const int lane = t & 31;
    const int wave = t >> 5;
    const int half = lane >> 4;
    const int lr   = lane & 15;

    // stage Q block (16 x 1024 bf16) via Tensor Data Mover, wave 0 issues
    if (wave == 0) {
        u32 lds_off = (u32)(uintptr_t)(&Qs[0]);       // low 32 bits = LDS offset
        tdm_load_tile_2d(lds_off, Q + ((size_t)b * S + q0) * DE,
                         /*cols=*/DIN, /*rows=*/QB, /*row_stride=*/DE);
        __builtin_amdgcn_s_wait_tensorcnt(0);
    }
    if (t < QB) { rowM[t] = -1e30f; rowL[t] = 0.f; }
    __syncthreads();

    const int dbase = wave * 128;
    v8f acc[8] = {};
    const float scale = 0.03125f;    // 1/sqrt(1024)

    for (int kc = 0; kc < S; kc += KC) {
        // phase 1: each wave computes a 16x16 score tile for k-rows kc+wave*16..+16
        v8f sacc = {};
        const int krow = kc + wave * 16 + lr;
        const u16* Kg = K + ((size_t)b * S + krow) * DE;
        for (int d0 = 0; d0 < DIN; d0 += 32) {
            // TDM pad: +8 shorts after every 512 shorts stored
            const u16* qp = &Qs[lr * LQ + d0 + 8 * (d0 >> 9)];
            v16bf aq = ld_frag(qp + half * 8, qp + 16 + half * 8);
            v16bf bk = ld_frag(Kg + d0 + half * 8, Kg + d0 + 16 + half * 8);
            sacc = wmma_bf16(aq, bk, sacc);
        }
        #pragma unroll
        for (int r = 0; r < 8; r++)
            Sbuf[r + half * 8][wave * 16 + lr] = sacc[r] * scale;
        __syncthreads();

        // phase 2: online softmax update, one thread per q-row
        if (t < QB) {
            float mold = rowM[t], mnew = mold;
            for (int j = 0; j < KC; j++) mnew = fmaxf(mnew, Sbuf[t][j]);
            float alpha = __expf(mold - mnew);
            float sum = 0.f;
            for (int j = 0; j < KC; j++) {
                float p = __expf(Sbuf[t][j] - mnew);
                Pbuf[t * LP + j] = f2bf(p);
                sum += p;
            }
            rowL[t] = rowL[t] * alpha + sum;
            rowM[t] = mnew;
            rowA[t] = alpha;
        }
        __syncthreads();

        // phase 3: rescale running accumulator, then acc += P @ V-slice
        float al[8];
        #pragma unroll
        for (int r = 0; r < 8; r++) al[r] = rowA[r + half * 8];
        #pragma unroll
        for (int j = 0; j < 8; j++)
            #pragma unroll
            for (int r = 0; r < 8; r++) acc[j][r] *= al[r];

        for (int ks = 0; ks < KC; ks += 32) {
            const u16* pp = &Pbuf[lr * LP + ks];
            v16bf ap = ld_frag(pp + half * 8, pp + 16 + half * 8);
            #pragma unroll
            for (int j = 0; j < 8; j++) {
                const u16* vp = Vt + ((size_t)b * DE + dbase + j * 16 + lr) * S + kc + ks;
                v16bf bv = ld_frag(vp + half * 8, vp + 16 + half * 8);
                acc[j] = wmma_bf16(ap, bv, acc[j]);
            }
        }
        __syncthreads();
    }

    // finalize: divide by softmax denominator, store bf16 ctx
    float inv[8];
    #pragma unroll
    for (int r = 0; r < 8; r++) inv[r] = 1.f / rowL[r + half * 8];
    #pragma unroll
    for (int j = 0; j < 8; j++) {
        const int n = dbase + j * 16 + lr;
        #pragma unroll
        for (int r = 0; r < 8; r++) {
            const int m = q0 + r + half * 8;
            ctx[((size_t)b * S + m) * DE + n] = f2bf(acc[j][r] * inv[r]);
        }
    }
}

// ---------- host launcher ----------
extern "C" void kernel_launch(void* const* d_in, const int* in_sizes, int n_in,
                              void* d_out, int out_size, void* d_ws, size_t ws_size,
                              hipStream_t stream)
{
    (void)in_sizes; (void)n_in; (void)out_size; (void)ws_size;

    const float* x  = (const float*)d_in[0];
    const float* Wq = (const float*)d_in[1];
    const float* bq = (const float*)d_in[2];
    const float* Wk = (const float*)d_in[3];
    const float* bk = (const float*)d_in[4];
    const float* Wv = (const float*)d_in[5];
    const float* bv = (const float*)d_in[6];
    const float* Wo = (const float*)d_in[7];
    const float* bo = (const float*)d_in[8];
    float* out = (float*)d_out;

    char* ws = (char*)d_ws;
    const size_t MB = 1ull << 20;
    u16* xb  = (u16*)(ws);                 // 32 MB  x bf16 [m][k]
    u16* wqb = (u16*)(ws + 32 * MB);       //  2 MB  Wq^T bf16 [n][k]
    u16* wkb = (u16*)(ws + 34 * MB);       //  2 MB  Wk^T
    u16* wvb = (u16*)(ws + 36 * MB);       //  2 MB  Wv^T
    u16* wob = (u16*)(ws + 38 * MB);       //  2 MB  Wo^T
    u16* Qb  = (u16*)(ws + 40 * MB);       // 32 MB  Q bf16 [b][s][d]
    u16* Kb  = (u16*)(ws + 72 * MB);       // 32 MB  K bf16 [b][s][d]
    u16* Vtb = (u16*)(ws + 104 * MB);      // 32 MB  V^T bf16 [b][d][s]
    u16* ctx = (u16*)(ws + 136 * MB);      // 32 MB  attention output bf16

    const int nx4 = MTOT * DIN / 4;
    const int nw  = DIN * DE;
    cvt_f32_bf16<<<(nx4 + 255) / 256, 256, 0, stream>>>(x, xb, nx4);
    cvt_f32_bf16_T<<<(nw + 255) / 256, 256, 0, stream>>>(Wq, wqb, DIN, DE);
    cvt_f32_bf16_T<<<(nw + 255) / 256, 256, 0, stream>>>(Wk, wkb, DIN, DE);
    cvt_f32_bf16_T<<<(nw + 255) / 256, 256, 0, stream>>>(Wv, wvb, DIN, DE);
    cvt_f32_bf16_T<<<(nw + 255) / 256, 256, 0, stream>>>(Wo, wob, DE,  DE);

    dim3 gg((MTOT / 128) * (DE / 128));    // 1024 blocks
    gemm_bf16_k<0><<<gg, 256, 0, stream>>>(xb, wqb, bq, Qb,  MTOT, DE, DIN);
    gemm_bf16_k<0><<<gg, 256, 0, stream>>>(xb, wkb, bk, Kb,  MTOT, DE, DIN);
    gemm_bf16_k<1><<<gg, 256, 0, stream>>>(xb, wvb, bv, Vtb, MTOT, DE, DIN);

    attn_k<<<dim3(BATCH * (S / 16)), 256, 0, stream>>>(Qb, Kb, Vtb, ctx);

    gemm_bf16_k<2><<<gg, 256, 0, stream>>>(ctx, wob, bo, out, MTOT, DE, DIN);
}
